// SparseLiftAttention_66314295050801
// MI455X (gfx1250) — compile-verified, matrix operand order checked by reading, not run on hardware
//
#include <hip/hip_runtime.h>
#include <hip/hip_bf16.h>
#include <math.h>

// ---------------------------------------------------------------------------
// SparseLiftAttention for MI455X (gfx1250, wave32, WMMA bf16 16x16x32).
// Pipeline: [gemm Q][gemm K][gemm V] -> [topk32 Q][topk32 K] -> [flash attn]
//           -> [gemm W_o].  All matmuls via v_wmma_f32_16x16x32_bf16.
// Compute-bound (~45 GFLOP, ~100MB working set fits in 192MB L2), so all
// GEMM-shaped work runs on WMMA with double-buffered LDS staging to hide
// global-load latency behind the WMMA phase.
// ---------------------------------------------------------------------------

typedef __attribute__((ext_vector_type(16))) __bf16 v16bf;
typedef __attribute__((ext_vector_type(8)))  __bf16 v8bf;
typedef __attribute__((ext_vector_type(8)))  float  v8f;

#define T_DIM 2048
#define D_DIM 1024
#define H_DIM 16
#define HD    64
#define NL    128   // lifted dims per head
#define TK    32

static __device__ inline __bf16 f2bf(float f) { return (__bf16)f; }

static __device__ inline v16bf cat16(v8bf a, v8bf b) {
  return __builtin_shufflevector(a, b, 0,1,2,3,4,5,6,7,8,9,10,11,12,13,14,15);
}

// load 8 contiguous f32 from global, convert to bf16 vector
static __device__ inline v8bf cvt8(const float* p) {
  const float4 f0 = *(const float4*)p;
  const float4 f1 = *(const float4*)(p + 4);
  v8bf r;
  r[0] = f2bf(f0.x); r[1] = f2bf(f0.y); r[2] = f2bf(f0.z); r[3] = f2bf(f0.w);
  r[4] = f2bf(f1.x); r[5] = f2bf(f1.y); r[6] = f2bf(f1.z); r[7] = f2bf(f1.w);
  return r;
}

static __device__ inline void st4bf(__bf16* d, float4 v) {
  d[0] = f2bf(v.x); d[1] = f2bf(v.y); d[2] = f2bf(v.z); d[3] = f2bf(v.w);
}

// ---------------------------------------------------------------------------
// C(MxN) = A(MxK) @ W(NxK)^T, f32 in/out, bf16 WMMA compute. N,K compile-time
// so strides constant-fold (immediate-offset stores, no u64 add chains).
// Block: 256 threads = 8 waves; tile 128(M) x 128(N); K stepped by 32 through
// double-buffered LDS so next tile's global loads overlap the 64 WMMAs.
// ---------------------------------------------------------------------------
template <int N, int K>
__global__ __launch_bounds__(256) void gemm_xwt_kernel(
    const float* __restrict__ A, const float* __restrict__ W,
    float* __restrict__ C)
{
  __shared__ __bf16 Asm[2][128 * 40];  // 128 rows x 32 k, stride 40 (16B align)
  __shared__ __bf16 Bsm[2][128 * 40];

  const int tid  = threadIdx.x;
  const int wave = tid >> 5;
  const int lane = tid & 31;
  const int hl   = lane >> 4;
  const int ln   = lane & 15;

  const int rowb = blockIdx.y * 128;
  const int colb = blockIdx.x * 128;

  // per-thread staging coords: 8 float4 per 32-wide row
  const int sr = tid >> 3;        // row 0..31 (+32*i)
  const int sc = (tid & 7) * 4;   // col within 32

  const v8f zf = {0.f,0.f,0.f,0.f,0.f,0.f,0.f,0.f};
  v8f acc[8];
#pragma unroll
  for (int nt = 0; nt < 8; ++nt) acc[nt] = zf;

  float4 pa[4], pb[4];
  // prefetch k-tile 0
#pragma unroll
  for (int i = 0; i < 4; ++i) {
    pa[i] = *(const float4*)(A + (long)(rowb + sr + 32 * i) * K + sc);
    pb[i] = *(const float4*)(W + (long)(colb + sr + 32 * i) * K + sc);
  }

  int cur = 0;
  for (int k0 = 0; k0 < K; k0 += 32) {
    // convert + store prefetched tile into LDS[cur]
#pragma unroll
    for (int i = 0; i < 4; ++i) {
      st4bf(&Asm[cur][(sr + 32 * i) * 40 + sc], pa[i]);
      st4bf(&Bsm[cur][(sr + 32 * i) * 40 + sc], pb[i]);
    }
    __syncthreads();
    // issue next tile's global loads; their wait hides behind the WMMAs
    if (k0 + 32 < K) {
#pragma unroll
      for (int i = 0; i < 4; ++i) {
        pa[i] = *(const float4*)(A + (long)(rowb + sr + 32 * i) * K + k0 + 32 + sc);
        pb[i] = *(const float4*)(W + (long)(colb + sr + 32 * i) * K + k0 + 32 + sc);
      }
    }
    // A fragment: lane = row (wave*16+ln), K pattern {hl*8+0..7, 16+hl*8+0..7}
    const __bf16* ar = &Asm[cur][(wave * 16 + ln) * 40 + hl * 8];
    const v16bf aF = cat16(*(const v8bf*)ar, *(const v8bf*)(ar + 16));
#pragma unroll
    for (int nt = 0; nt < 8; ++nt) {
      const __bf16* br = &Bsm[cur][(nt * 16 + ln) * 40 + hl * 8];
      const v16bf bF = cat16(*(const v8bf*)br, *(const v8bf*)(br + 16));
      acc[nt] = __builtin_amdgcn_wmma_f32_16x16x32_bf16(
          false, aF, false, bF, (short)0, acc[nt], false, false);
    }
    cur ^= 1;
  }

  // store C: base + compile-time immediate offsets (j*N + nt*16)
  float* cb = C + (long)(rowb + wave * 16 + hl * 8) * N + colb + ln;
#pragma unroll
  for (int nt = 0; nt < 8; ++nt) {
#pragma unroll
    for (int j = 0; j < 8; ++j) cb[j * N + nt * 16] = acc[nt][j];
  }
}

// ---------------------------------------------------------------------------
// ReLU + keep top-32 of each 128-wide row (in place). One wave per row.
// Iterative wave-max; killing all ties is exact a.s. for continuous data,
// and degenerates to plain ReLU when a row has <32 positive entries.
// ---------------------------------------------------------------------------
__global__ __launch_bounds__(256) void topk32_kernel(float* __restrict__ buf)
{
  const int wave = threadIdx.x >> 5;
  const int lane = threadIdx.x & 31;
  const long row = (long)blockIdx.x * 8 + wave;
  float* p = buf + row * NL;

  float o[4], v[4];
#pragma unroll
  for (int s = 0; s < 4; ++s) {
    float x = p[s * 32 + lane];
    x = x > 0.f ? x : 0.f;   // relu
    o[s] = x; v[s] = x;
  }

  float t = -1.0f;
  for (int it = 0; it < TK; ++it) {
    float m = fmaxf(fmaxf(v[0], v[1]), fmaxf(v[2], v[3]));
#pragma unroll
    for (int off = 16; off; off >>= 1) m = fmaxf(m, __shfl_xor(m, off, 32));
    t = m;   // after 32 iters: 32nd largest (or -1 => keep-all fallback)
#pragma unroll
    for (int s = 0; s < 4; ++s) if (v[s] == m) v[s] = -1.0f;
  }

#pragma unroll
  for (int s = 0; s < 4; ++s) p[s * 32 + lane] = (o[s] >= t) ? o[s] : 0.0f;
}

// ---------------------------------------------------------------------------
// Flash-style causal p1-normalized attention, one head per blockIdx.y.
// Block = 8 waves x 16 query rows = 128 queries. Key tiles of 32, K/V staged
// through double-buffered LDS with register prefetch.
// St = K(32x128) @ Q^T per tile; the C-layout of St IS the A-layout of S
// after a per-lane bf16 pack, so S@V needs no cross-lane transpose.
// Denominator via one WMMA against an all-ones B matrix (same C layout as Y).
// ---------------------------------------------------------------------------
__global__ __launch_bounds__(256) void attn_kernel(
    const float* __restrict__ qs, const float* __restrict__ ks,
    const float* __restrict__ vv, const float* __restrict__ sink,
    const float* __restrict__ logbeta, float* __restrict__ yh)
{
  __shared__ __bf16 Ksm[2][32 * 136];  // 32 keys x 128 lifted dims (pad->136)
  __shared__ __bf16 Vtm[2][64 * 40];   // V transposed: 64 d x 32 keys (pad->40)

  const int h    = blockIdx.y;
  const int qblk = blockIdx.x * 128;
  const int tid  = threadIdx.x;
  const int wave = tid >> 5;
  const int lane = tid & 31;
  const int hl   = lane >> 4;
  const int ln   = lane & 15;
  const int qw   = qblk + wave * 16;   // this wave's first query row

  // staging coords
  const int kr  = tid >> 5;            // K: row 0..7 (+8*i), 32 f4 per row
  const int kc  = (tid & 31) * 4;
  const int vr  = tid >> 4;            // V: row 0..15 (+16*i), 16 f4 per row
  const int vc  = (tid & 15) * 4;

  const float beta = __expf(logbeta[0]);
  const v8f zf = {0.f,0.f,0.f,0.f,0.f,0.f,0.f,0.f};

  v16bf ones;
#pragma unroll
  for (int i = 0; i < 16; ++i) ones[i] = f2bf(1.0f);

  // Q fragments (B operand for St = K @ Q^T): lane = query col qw+ln
  v16bf qB[4];
  {
    const float* qrow = qs + (long)(qw + ln) * (H_DIM * NL) + h * NL;
#pragma unroll
    for (int f = 0; f < 4; ++f)
      qB[f] = cat16(cvt8(qrow + f * 32 + hl * 8),
                    cvt8(qrow + f * 32 + 16 + hl * 8));
  }

  v8f yacc[4];
#pragma unroll
  for (int nt = 0; nt < 4; ++nt) yacc[nt] = zf;
  v8f dacc = zf;

  const int nkt = qblk / 32 + 4;   // causal bound for this block

  float4 pk[4], pv[2];
#pragma unroll
  for (int i = 0; i < 4; ++i)
    pk[i] = *(const float4*)(ks + (long)(kr + 8 * i) * (H_DIM * NL) + h * NL + kc);
#pragma unroll
  for (int i = 0; i < 2; ++i)
    pv[i] = *(const float4*)(vv + (long)(vr + 16 * i) * (H_DIM * HD) + h * HD + vc);

  int cur = 0;
  for (int kt = 0; kt < nkt; ++kt) {
    const int kb = kt * 32;
    // convert + store prefetched K/V tile into LDS[cur]
#pragma unroll
    for (int i = 0; i < 4; ++i)
      st4bf(&Ksm[cur][(kr + 8 * i) * 136 + kc], pk[i]);
#pragma unroll
    for (int i = 0; i < 2; ++i) {
      const int key = vr + 16 * i;
      Vtm[cur][(vc + 0) * 40 + key] = f2bf(pv[i].x);
      Vtm[cur][(vc + 1) * 40 + key] = f2bf(pv[i].y);
      Vtm[cur][(vc + 2) * 40 + key] = f2bf(pv[i].z);
      Vtm[cur][(vc + 3) * 40 + key] = f2bf(pv[i].w);
    }
    __syncthreads();
    // prefetch next key tile (global loads overlap the WMMA phase)
    if (kt + 1 < nkt) {
      const int nb = kb + 32;
#pragma unroll
      for (int i = 0; i < 4; ++i)
        pk[i] = *(const float4*)(ks + (long)(nb + kr + 8 * i) * (H_DIM * NL) + h * NL + kc);
#pragma unroll
      for (int i = 0; i < 2; ++i)
        pv[i] = *(const float4*)(vv + (long)(nb + vr + 16 * i) * (H_DIM * HD) + h * HD + vc);
    }

    if (kb <= qw + 15) {  // causal: this wave still has keys in range
      // St = K(32x128) @ Q^T(128x16): two 16x16 C tiles, 4 k-steps of 32
      v8f st0 = zf, st1 = zf;
#pragma unroll
      for (int f = 0; f < 4; ++f) {
        const __bf16* kr0 = &Ksm[cur][ln * 136 + f * 32 + hl * 8];
        const v16bf kA0 = cat16(*(const v8bf*)kr0, *(const v8bf*)(kr0 + 16));
        st0 = __builtin_amdgcn_wmma_f32_16x16x32_bf16(
            false, kA0, false, qB[f], (short)0, st0, false, false);
        const __bf16* kr1 = &Ksm[cur][(16 + ln) * 136 + f * 32 + hl * 8];
        const v16bf kA1 = cat16(*(const v8bf*)kr1, *(const v8bf*)(kr1 + 16));
        st1 = __builtin_amdgcn_wmma_f32_16x16x32_bf16(
            false, kA1, false, qB[f], (short)0, st1, false, false);
      }

      // causal mask + pack St (C layout) into S A-fragment (bf16), same lane
      const int qg = qw + ln;
      v16bf sA;
#pragma unroll
      for (int j = 0; j < 8; ++j) {
        const int kg0 = kb + hl * 8 + j;
        const int kg1 = kb + 16 + hl * 8 + j;
        sA[j]     = f2bf((kg0 <= qg) ? st0[j] : 0.f);
        sA[j + 8] = f2bf((kg1 <= qg) ? st1[j] : 0.f);
      }

      // denom += S @ ones ; Y += S @ V
      dacc = __builtin_amdgcn_wmma_f32_16x16x32_bf16(
          false, sA, false, ones, (short)0, dacc, false, false);
#pragma unroll
      for (int nt = 0; nt < 4; ++nt) {
        const __bf16* vp = &Vtm[cur][(nt * 16 + ln) * 40 + hl * 8];
        const v16bf vB = cat16(*(const v8bf*)vp, *(const v8bf*)(vp + 16));
        yacc[nt] = __builtin_amdgcn_wmma_f32_16x16x32_bf16(
            false, sA, false, vB, (short)0, yacc[nt], false, false);
      }
    }
    cur ^= 1;
  }

  // normalize: Y = (acc + beta*sink_d) / max(denom + beta, eps)
  const float* sk = sink + h * HD;
  float* yb = yh + (long)(qw + hl * 8) * (H_DIM * HD) + h * HD + ln;
#pragma unroll
  for (int nt = 0; nt < 4; ++nt) {
    const float skv = sk[nt * 16 + ln];
#pragma unroll
    for (int j = 0; j < 8; ++j) {
      const float den = fmaxf(dacc[j] + beta, 1e-12f);
      yb[(long)j * (H_DIM * HD) + nt * 16] = (yacc[nt][j] + beta * skv) / den;
    }
  }
}

// ---------------------------------------------------------------------------
extern "C" void kernel_launch(void* const* d_in, const int* in_sizes, int n_in,
                              void* d_out, int out_size, void* d_ws, size_t ws_size,
                              hipStream_t stream) {
  (void)in_sizes; (void)n_in; (void)out_size; (void)ws_size;
  const float* x    = (const float*)d_in[0];
  const float* Wq   = (const float*)d_in[1];
  const float* Wk   = (const float*)d_in[2];
  const float* Wv   = (const float*)d_in[3];
  const float* Wo   = (const float*)d_in[4];
  const float* sink = (const float*)d_in[5];
  const float* lb   = (const float*)d_in[6];
  float* out = (float*)d_out;

  float* q  = (float*)d_ws;                       // T x (H*NL)  = 2048x2048
  float* k  = q + (size_t)T_DIM * (H_DIM * NL);   // 2048x2048
  float* v  = k + (size_t)T_DIM * (H_DIM * NL);   // 2048x1024
  float* yh = v + (size_t)T_DIM * (H_DIM * HD);   // 2048x1024

  const dim3 blk(256);
  // projections: q = x Wq^T, k = x Wk^T, v = x Wv^T
  gemm_xwt_kernel<H_DIM * NL, D_DIM><<<dim3(16, 16), blk, 0, stream>>>(x, Wq, q);
  gemm_xwt_kernel<H_DIM * NL, D_DIM><<<dim3(16, 16), blk, 0, stream>>>(x, Wk, k);
  gemm_xwt_kernel<H_DIM * HD, D_DIM><<<dim3(8, 16),  blk, 0, stream>>>(x, Wv, v);
  // relu + top-32 sparsify q, k in place (32768 rows each, 8 rows/block)
  topk32_kernel<<<dim3((T_DIM * H_DIM) / 8), blk, 0, stream>>>(q);
  topk32_kernel<<<dim3((T_DIM * H_DIM) / 8), blk, 0, stream>>>(k);
  // causal p1 attention with sink mass
  attn_kernel<<<dim3(T_DIM / 128, H_DIM), blk, 0, stream>>>(q, k, v, sink, lb, yh);
  // output projection
  gemm_xwt_kernel<D_DIM, H_DIM * HD><<<dim3(8, 16), blk, 0, stream>>>(yh, Wo, out);
}
